// StableDynamicsModel_27376121545312
// MI455X (gfx1250) — matrix operand order repaired and checked by reading, not compiled, not run on hardware
//
#include <hip/hip_runtime.h>

#define B_ROWS 32768
#define D_DIM  512
#define H_DIM  64
#define ALPHA_C 0.9f
#define EPS_C   1e-3f

typedef __attribute__((ext_vector_type(16))) __bf16       v16bf;
typedef __attribute__((ext_vector_type(2)))  __bf16       v2bf;
typedef __attribute__((ext_vector_type(8)))  float        v8f;
typedef __attribute__((ext_vector_type(8)))  unsigned int v8u;

// ---------- scalar helpers (native bf16 converts) ----------
__device__ __forceinline__ unsigned short f2bf(float f) {
  __bf16 h = (__bf16)f;                      // native v_cvt f32->bf16
  return __builtin_bit_cast(unsigned short, h);
}
__device__ __forceinline__ unsigned int packbf2(float a, float b) {
  v2bf t;
  t[0] = (__bf16)a;
  t[1] = (__bf16)b;                          // should fuse to v_cvt_pk_bf16_f32
  return __builtin_bit_cast(unsigned int, t);
}
__device__ __forceinline__ float bf2f(unsigned short h) {
  return __uint_as_float(((unsigned int)h) << 16);
}
__device__ __forceinline__ v8f zero8() {
  v8f z;
#pragma unroll
  for (int i = 0; i < 8; ++i) z[i] = 0.0f;
  return z;
}
__device__ __forceinline__ v8f wmma_bf16(v16bf a, v16bf b, v8f c) {
  // D = A(16x32 bf16) * B(32x16 bf16) + C(16x16 f32)
  return __builtin_amdgcn_wmma_f32_16x16x32_bf16(false, a, false, b, (short)0, c,
                                                 false, false);
}
__device__ __forceinline__ v16bf pack2(uint4 x0, uint4 x1) {
  v8u u;
  u[0] = x0.x; u[1] = x0.y; u[2] = x0.z; u[3] = x0.w;
  u[4] = x1.x; u[5] = x1.y; u[6] = x1.z; u[7] = x1.w;
  return __builtin_bit_cast(v16bf, u);
}

// ---------- WMMA operand loaders (wave32 layouts per CDNA5 ISA 7.12.2) ----------
// A operand 16x32 bf16 from row-major ushort tile [16 rows][stride], K window k0.
// lane<16: row=lane, K = k0+[0..7] and k0+16+[0..7]; lane>=16: +8 on both segments.
__device__ __forceinline__ v16bf loadA_u16(const unsigned short* base, int stride, int k0) {
  int l = threadIdx.x & 31;
  const unsigned short* p = base + (l & 15) * stride + k0 + ((l & 16) ? 8 : 0);
  return pack2(*(const uint4*)p, *(const uint4*)(p + 16));
}
// B operand 32x16 bf16 from row-major ushort [N][stride=K], cols n0.., K window k0.
// lane<16: col=n0+lane, K = k0+[0..7] and k0+16+[0..7]; lane>=16: +8 on both segments.
__device__ __forceinline__ v16bf loadB_u16(const unsigned short* base, int stride, int n0, int k0) {
  int l = threadIdx.x & 31;
  const unsigned short* p = base + (n0 + (l & 15)) * stride + k0 + ((l & 16) ? 8 : 0);
  return pack2(*(const uint4*)p, *(const uint4*)(p + 16));
}
// A operand from f32 row-major source, converting to bf16 with native pack converts.
__device__ __forceinline__ v16bf loadA_f32(const float* base, int stride, int k0) {
  int l = threadIdx.x & 31;
  const float* p = base + (l & 15) * stride + k0 + ((l & 16) ? 8 : 0);
  float4 f0 = *(const float4*)p;
  float4 f1 = *(const float4*)(p + 4);
  float4 f2 = *(const float4*)(p + 16);
  float4 f3 = *(const float4*)(p + 20);
  v8u u;
  u[0] = packbf2(f0.x, f0.y);  u[1] = packbf2(f0.z, f0.w);
  u[2] = packbf2(f1.x, f1.y);  u[3] = packbf2(f1.z, f1.w);
  u[4] = packbf2(f2.x, f2.y);  u[5] = packbf2(f2.z, f2.w);
  u[6] = packbf2(f3.x, f3.y);  u[7] = packbf2(f3.z, f3.w);
  return __builtin_bit_cast(v16bf, u);
}

// 16x64 = relu( A(16x64 from LDS) @ Bw(64x64, [N][K] layout) + bias ) -> LDS bf16
__device__ __forceinline__ void layer64_relu(const unsigned short* Asrc,
                                             const unsigned short* Bw,
                                             const float* __restrict__ bias,
                                             unsigned short* dst) {
  int lane = threadIdx.x & 31;
  int ccol = lane & 15, crow0 = (lane >> 4) << 3;
  v8f acc[4] = {zero8(), zero8(), zero8(), zero8()};
#pragma unroll
  for (int ks = 0; ks < 64; ks += 32) {
    v16bf a = loadA_u16(Asrc, 64, ks);
#pragma unroll
    for (int t = 0; t < 4; ++t)
      acc[t] = wmma_bf16(a, loadB_u16(Bw, 64, t * 16, ks), acc[t]);
  }
#pragma unroll
  for (int t = 0; t < 4; ++t) {
    float bb = bias[t * 16 + ccol];
#pragma unroll
    for (int r = 0; r < 8; ++r)
      dst[(crow0 + r) * 64 + t * 16 + ccol] = f2bf(fmaxf(acc[t][r] + bb, 0.0f));
  }
}

// ---------- kernel 0: weights -> bf16, both orientations ----------
__global__ __launch_bounds__(256) void k_prep(
    const float* __restrict__ W1, const float* __restrict__ W2, const float* __restrict__ W3,
    const float* __restrict__ V1, const float* __restrict__ V2,
    unsigned short* __restrict__ W1c, unsigned short* __restrict__ W1t,
    unsigned short* __restrict__ W2c, unsigned short* __restrict__ W2t,
    unsigned short* __restrict__ W3c, unsigned short* __restrict__ W3t,
    unsigned short* __restrict__ V1c, unsigned short* __restrict__ V1t,
    unsigned short* __restrict__ V2c, unsigned short* __restrict__ V2t) {
  int tid = blockIdx.x * blockDim.x + threadIdx.x;
  int n = gridDim.x * blockDim.x;
  for (int i = tid; i < 512 * 64; i += n) {      // W1, V1: [512][64]
    int d = i >> 6, h = i & 63;
    unsigned short w = f2bf(W1[i]);
    W1c[i] = w;  W1t[h * 512 + d] = w;
    unsigned short v = f2bf(V1[i]);
    V1c[i] = v;  V1t[h * 512 + d] = v;
  }
  for (int i = tid; i < 64 * 512; i += n) {      // W3: [64][512]
    int h = i >> 9, d = i & 511;
    unsigned short w = f2bf(W3[i]);
    W3c[i] = w;  W3t[d * 64 + h] = w;
  }
  for (int i = tid; i < 64 * 64; i += n) {       // W2, V2: [64][64]
    int a = i >> 6, b = i & 63;
    unsigned short w = f2bf(W2[i]);
    W2c[i] = w;  W2t[b * 64 + a] = w;
    unsigned short v = f2bf(V2[i]);
    V2c[i] = v;  V2t[b * 64 + a] = v;
  }
}

// ---------- kernel 1: fused fwd+bwd, one wave per 16-row slab ----------
__global__ __launch_bounds__(256) void k_fused(
    const float* __restrict__ x,
    const float* __restrict__ b1, const float* __restrict__ b2, const float* __restrict__ b3,
    const float* __restrict__ c1, const float* __restrict__ c2,
    const float* __restrict__ V3, const float* __restrict__ c3,
    const unsigned short* __restrict__ W1c, const unsigned short* __restrict__ W1t,
    const unsigned short* __restrict__ W2c, const unsigned short* __restrict__ W2t,
    const unsigned short* __restrict__ W3c, const unsigned short* __restrict__ W3t,
    const unsigned short* __restrict__ V1c, const unsigned short* __restrict__ V1t,
    const unsigned short* __restrict__ V2c, const unsigned short* __restrict__ V2t,
    float* __restrict__ fout,   // d_out: holds f after this kernel
    float* __restrict__ gv,     // ws: grad_v [B,512]
    float* __restrict__ dotb,   // ws: per-row sum gv^2
    float* __restrict__ sb)     // ws: relu(dot + alpha*lyap)
{
  __shared__ __align__(16) unsigned short smem[8 * 6 * 1024];
  __shared__ float slyap[8 * 16];

  const int wave = threadIdx.x >> 5;
  const int lane = threadIdx.x & 31;
  const int m0 = (blockIdx.x * 8 + wave) * 16;
  const int ccol = lane & 15;
  const int crow0 = (lane >> 4) << 3;

  unsigned short* S    = smem + wave * 6 * 1024;   // wave-private LDS tiles (16x64 bf16)
  unsigned short* S_H1 = S;
  unsigned short* S_H2 = S + 1024;
  unsigned short* S_G1 = S + 2048;
  unsigned short* S_G2 = S + 3072;
  unsigned short* S_TMP= S + 4096;
  unsigned short* S_DP = S + 5120;
  float* lyapw = slyap + wave * 16;

  // ---- h1 = relu(x @ W1 + b1), K=512 streamed from global f32 x ----
  {
    v8f acc[4] = {zero8(), zero8(), zero8(), zero8()};
#pragma unroll 4
    for (int kk = 0; kk < 512; kk += 32) {
      v16bf a = loadA_f32(x + (size_t)m0 * 512, 512, kk);
#pragma unroll
      for (int t = 0; t < 4; ++t)
        acc[t] = wmma_bf16(a, loadB_u16(W1t, 512, t * 16, kk), acc[t]);
    }
#pragma unroll
    for (int t = 0; t < 4; ++t) {
      float bb = b1[t * 16 + ccol];
#pragma unroll
      for (int r = 0; r < 8; ++r)
        S_H1[(crow0 + r) * 64 + t * 16 + ccol] = f2bf(fmaxf(acc[t][r] + bb, 0.0f));
    }
  }
  // ---- h2 = relu(h1 @ W2 + b2) ----
  layer64_relu(S_H1, W2t, b2, S_H2);

  // ---- f = h2 @ W3 + b3 (chunks of 64 cols): write f32 to d_out, stage bf16,
  //      accumulate g1pre += f_chunk @ V1, accumulate eps-norm ----
  float ns[8];
#pragma unroll
  for (int r = 0; r < 8; ++r) ns[r] = 0.0f;
  v8f g1acc[4] = {zero8(), zero8(), zero8(), zero8()};
  for (int j = 0; j < 8; ++j) {
    v8f acc[4] = {zero8(), zero8(), zero8(), zero8()};
#pragma unroll
    for (int ks = 0; ks < 64; ks += 32) {
      v16bf a = loadA_u16(S_H2, 64, ks);
#pragma unroll
      for (int t = 0; t < 4; ++t)
        acc[t] = wmma_bf16(a, loadB_u16(W3t, 64, j * 64 + t * 16, ks), acc[t]);
    }
#pragma unroll
    for (int t = 0; t < 4; ++t) {
      float bb = b3[j * 64 + t * 16 + ccol];
#pragma unroll
      for (int r = 0; r < 8; ++r) {
        int row = crow0 + r;
        float v = acc[t][r] + bb;
        fout[(size_t)(m0 + row) * 512 + j * 64 + t * 16 + ccol] = v;
        ns[r] += v * v;
        S_TMP[row * 64 + t * 16 + ccol] = f2bf(v);
      }
    }
#pragma unroll
    for (int ks = 0; ks < 64; ks += 32) {
      v16bf a = loadA_u16(S_TMP, 64, ks);
#pragma unroll
      for (int t = 0; t < 4; ++t)
        g1acc[t] = wmma_bf16(a, loadB_u16(V1t, 512, t * 16, j * 64 + ks), g1acc[t]);
    }
  }
  // ---- g1 = relu(g1acc + c1); g2 = relu(g1 @ V2 + c2) ----
#pragma unroll
  for (int t = 0; t < 4; ++t) {
    float bb = c1[t * 16 + ccol];
#pragma unroll
    for (int r = 0; r < 8; ++r)
      S_G1[(crow0 + r) * 64 + t * 16 + ccol] = f2bf(fmaxf(g1acc[t][r] + bb, 0.0f));
  }
  layer64_relu(S_G1, V2t, c2, S_G2);

  // ---- per-row ||f||^2 reduce, then lyap = g2.V3 + c3 + eps*||f||^2 ----
#pragma unroll
  for (int r = 0; r < 8; ++r) {
    float v = ns[r];
    v += __shfl_xor(v, 1);  v += __shfl_xor(v, 2);
    v += __shfl_xor(v, 4);  v += __shfl_xor(v, 8);
    ns[r] = v;
  }
  if (ccol == 0) {
#pragma unroll
    for (int r = 0; r < 8; ++r) lyapw[crow0 + r] = ns[r];
  }
  if (lane < 16) {
    float acc = 0.0f;
    for (int k = 0; k < 64; ++k) acc += bf2f(S_G2[lane * 64 + k]) * V3[k];
    lyapw[lane] = acc + c3[0] + EPS_C * lyapw[lane];
  }

  // ---- backward: dpre2 = (g2>0) * V3 ----
  for (int i = 0; i < 32; ++i) {
    int e = lane + i * 32;
    S_TMP[e] = f2bf((S_G2[e] != 0) ? V3[e & 63] : 0.0f);
  }
  // ---- dpre1 = (g1>0) * (dpre2 @ V2^T) ----
  {
    v8f acc[4] = {zero8(), zero8(), zero8(), zero8()};
#pragma unroll
    for (int ks = 0; ks < 64; ks += 32) {
      v16bf a = loadA_u16(S_TMP, 64, ks);
#pragma unroll
      for (int t = 0; t < 4; ++t)
        acc[t] = wmma_bf16(a, loadB_u16(V2c, 64, t * 16, ks), acc[t]);
    }
#pragma unroll
    for (int t = 0; t < 4; ++t)
#pragma unroll
      for (int r = 0; r < 8; ++r) {
        int row = crow0 + r, col = t * 16 + ccol;
        S_DP[row * 64 + col] = f2bf((S_G1[row * 64 + col] != 0) ? acc[t][r] : 0.0f);
      }
  }
  // ---- dz = dpre1 @ V1^T + 2*eps*f (chunks), accumulate dh2pre += dz @ W3^T ----
  v8f dh2acc[4] = {zero8(), zero8(), zero8(), zero8()};
  for (int j = 0; j < 8; ++j) {
    v8f acc[4] = {zero8(), zero8(), zero8(), zero8()};
#pragma unroll
    for (int ks = 0; ks < 64; ks += 32) {
      v16bf a = loadA_u16(S_DP, 64, ks);
#pragma unroll
      for (int t = 0; t < 4; ++t)
        acc[t] = wmma_bf16(a, loadB_u16(V1c, 64, j * 64 + t * 16, ks), acc[t]);
    }
#pragma unroll
    for (int t = 0; t < 4; ++t)
#pragma unroll
      for (int r = 0; r < 8; ++r) {
        int row = crow0 + r, col = j * 64 + t * 16 + ccol;
        float z = fout[(size_t)(m0 + row) * 512 + col];
        S_TMP[row * 64 + t * 16 + ccol] = f2bf(acc[t][r] + 2.0f * EPS_C * z);
      }
#pragma unroll
    for (int ks = 0; ks < 64; ks += 32) {
      v16bf a = loadA_u16(S_TMP, 64, ks);
#pragma unroll
      for (int t = 0; t < 4; ++t)
        dh2acc[t] = wmma_bf16(a, loadB_u16(W3c, 512, t * 16, j * 64 + ks), dh2acc[t]);
    }
  }
  // ---- dpre_h2 = (h2>0)*dh2acc ; dpre_h1 = (h1>0)*(dpre_h2 @ W2^T) ----
#pragma unroll
  for (int t = 0; t < 4; ++t)
#pragma unroll
    for (int r = 0; r < 8; ++r) {
      int row = crow0 + r, col = t * 16 + ccol;
      S_G2[row * 64 + col] = f2bf((S_H2[row * 64 + col] != 0) ? dh2acc[t][r] : 0.0f);
    }
  {
    v8f acc[4] = {zero8(), zero8(), zero8(), zero8()};
#pragma unroll
    for (int ks = 0; ks < 64; ks += 32) {
      v16bf a = loadA_u16(S_G2, 64, ks);
#pragma unroll
      for (int t = 0; t < 4; ++t)
        acc[t] = wmma_bf16(a, loadB_u16(W2c, 64, t * 16, ks), acc[t]);
    }
#pragma unroll
    for (int t = 0; t < 4; ++t)
#pragma unroll
      for (int r = 0; r < 8; ++r) {
        int row = crow0 + r, col = t * 16 + ccol;
        S_G1[row * 64 + col] = f2bf((S_H1[row * 64 + col] != 0) ? acc[t][r] : 0.0f);
      }
  }
  // ---- grad_v = dpre_h1 @ W1^T (chunks): store f32, accumulate per-row dot ----
  float dacc[8];
#pragma unroll
  for (int r = 0; r < 8; ++r) dacc[r] = 0.0f;
  for (int j = 0; j < 8; ++j) {
    v8f acc[4] = {zero8(), zero8(), zero8(), zero8()};
#pragma unroll
    for (int ks = 0; ks < 64; ks += 32) {
      v16bf a = loadA_u16(S_G1, 64, ks);
#pragma unroll
      for (int t = 0; t < 4; ++t)
        acc[t] = wmma_bf16(a, loadB_u16(W1c, 64, j * 64 + t * 16, ks), acc[t]);
    }
#pragma unroll
    for (int t = 0; t < 4; ++t)
#pragma unroll
      for (int r = 0; r < 8; ++r) {
        int row = crow0 + r;
        float v = acc[t][r];
        gv[(size_t)(m0 + row) * 512 + j * 64 + t * 16 + ccol] = v;
        dacc[r] += v * v;
      }
  }
#pragma unroll
  for (int r = 0; r < 8; ++r) {
    float v = dacc[r];
    v += __shfl_xor(v, 1);  v += __shfl_xor(v, 2);
    v += __shfl_xor(v, 4);  v += __shfl_xor(v, 8);
    dacc[r] = v;
  }
  if (ccol == 0) {
#pragma unroll
    for (int r = 0; r < 8; ++r) {
      int row = crow0 + r;
      dotb[m0 + row] = dacc[r];
      sb[m0 + row] = fmaxf(dacc[r] + ALPHA_C * lyapw[row], 0.0f);
    }
  }
}

// ---------- kernel 2: deterministic denom = sum(dot) ----------
__global__ __launch_bounds__(1024) void k_denom(const float* __restrict__ dotb,
                                                float* __restrict__ denom) {
  __shared__ float red[1024];
  float s = 0.0f;
  for (int i = threadIdx.x; i < B_ROWS; i += 1024) s += dotb[i];
  red[threadIdx.x] = s;
  __syncthreads();
  for (int st = 512; st > 0; st >>= 1) {
    if ((int)threadIdx.x < st) red[threadIdx.x] += red[threadIdx.x + st];
    __syncthreads();
  }
  if (threadIdx.x == 0) denom[0] = red[0];
}

// ---------- kernel 3: out = f - (s[b]/denom) * grad_v ----------
__global__ __launch_bounds__(256) void k_finalize(float* __restrict__ out,
                                                  const float* __restrict__ gvv,
                                                  const float* __restrict__ sb,
                                                  const float* __restrict__ denom) {
  size_t i = (size_t)blockIdx.x * blockDim.x + threadIdx.x;  // float4 index
  float inv = 1.0f / denom[0];
  float sc = sb[i >> 7] * inv;          // 128 float4 per 512-wide row
  const float4* g4 = (const float4*)gvv;
  float4* o4 = (float4*)out;
  float4 f = o4[i], g = g4[i];
  f.x -= sc * g.x; f.y -= sc * g.y; f.z -= sc * g.z; f.w -= sc * g.w;
  o4[i] = f;
}

extern "C" void kernel_launch(void* const* d_in, const int* in_sizes, int n_in,
                              void* d_out, int out_size, void* d_ws, size_t ws_size,
                              hipStream_t stream) {
  (void)in_sizes; (void)n_in; (void)out_size; (void)ws_size;
  const float* x  = (const float*)d_in[0];
  const float* W1 = (const float*)d_in[1];
  const float* b1 = (const float*)d_in[2];
  const float* W2 = (const float*)d_in[3];
  const float* b2 = (const float*)d_in[4];
  const float* W3 = (const float*)d_in[5];
  const float* b3 = (const float*)d_in[6];
  const float* V1 = (const float*)d_in[7];
  const float* c1 = (const float*)d_in[8];
  const float* V2 = (const float*)d_in[9];
  const float* c2 = (const float*)d_in[10];
  const float* V3 = (const float*)d_in[11];
  const float* c3 = (const float*)d_in[12];

  char* p = (char*)d_ws;
  size_t o = 0;
  auto take = [&](size_t nbytes) -> char* {
    char* r = p + o;
    o = (o + nbytes + 255) & ~(size_t)255;
    return r;
  };
  float* gvb   = (float*)take((size_t)B_ROWS * D_DIM * sizeof(float));
  float* dotb  = (float*)take((size_t)B_ROWS * sizeof(float));
  float* sbuf  = (float*)take((size_t)B_ROWS * sizeof(float));
  float* denom = (float*)take(sizeof(float));
  unsigned short* W1c = (unsigned short*)take(512 * 64 * 2);
  unsigned short* W1t = (unsigned short*)take(512 * 64 * 2);
  unsigned short* W2c = (unsigned short*)take(64 * 64 * 2);
  unsigned short* W2t = (unsigned short*)take(64 * 64 * 2);
  unsigned short* W3c = (unsigned short*)take(64 * 512 * 2);
  unsigned short* W3t = (unsigned short*)take(64 * 512 * 2);
  unsigned short* V1c = (unsigned short*)take(512 * 64 * 2);
  unsigned short* V1t = (unsigned short*)take(512 * 64 * 2);
  unsigned short* V2c = (unsigned short*)take(64 * 64 * 2);
  unsigned short* V2t = (unsigned short*)take(64 * 64 * 2);

  k_prep<<<64, 256, 0, stream>>>(W1, W2, W3, V1, V2,
                                 W1c, W1t, W2c, W2t, W3c, W3t, V1c, V1t, V2c, V2t);
  k_fused<<<B_ROWS / 128, 256, 0, stream>>>(x, b1, b2, b3, c1, c2, V3, c3,
                                            W1c, W1t, W2c, W2t, W3c, W3t,
                                            V1c, V1t, V2c, V2t,
                                            (float*)d_out, gvb, dotb, sbuf);
  k_denom<<<1, 1024, 0, stream>>>(dotb, denom);
  k_finalize<<<(B_ROWS * D_DIM / 4) / 256, 256, 0, stream>>>((float*)d_out, gvb, sbuf, denom);
}
